// CDVAEEncoder_7275674599865
// MI455X (gfx1250) — compile-verified, matrix operand order checked by reading, not run on hardware
//
#include <hip/hip_runtime.h>
#include <hip/hip_bf16.h>

// ---------------- problem constants (match reference) ----------------
#define N_NODES   4096
#define HDIM      512
#define EDIM      64
#define NLAYERS   6
#define E_TOT     126976           // 128 graphs * 32*31 edges
#define EPN       31               // edges per source node (contiguous)
#define NCHUNKS   16
#define NODES_PER_CHUNK 256
#define E_CHUNK   (NODES_PER_CHUNK * EPN)   // 7936 = 124 * 64
#define NGRAPHS   128
#define LATD      256

typedef __bf16 bf16_t;
typedef bf16_t v16bf __attribute__((ext_vector_type(16)));
typedef float  v8f   __attribute__((ext_vector_type(8)));
typedef unsigned short ushort_t;

__device__ __forceinline__ ushort_t f2bf(float x) {
  union { float f; unsigned u; } v; v.f = x;
  unsigned r = v.u + 0x7FFFu + ((v.u >> 16) & 1u);   // RNE
  return (ushort_t)(r >> 16);
}
__device__ __forceinline__ float bf2f(ushort_t b) {
  union { float f; unsigned u; } v; v.u = ((unsigned)b) << 16; return v.f;
}
__device__ __forceinline__ float siluf(float x) {
  return x * (1.0f / (1.0f + __expf(-x)));
}

// fragment load from an n-major LDS row (64 bf16): two contiguous b128 runs
__device__ __forceinline__ v16bf ldsFrag(const ushort_t* rowBase, int fh, int ksub) {
  union { v16bf v; uint4 q[2]; } f;
  f.q[0] = *(const uint4*)(rowBase + ksub + (fh << 3));
  f.q[1] = *(const uint4*)(rowBase + ksub + 16 + (fh << 3));
  return f.v;
}

// ---------------- WMMA GEMM: C = act(A @ Bt^T + bias) ----------------
// BM=64 BN=64 BK=64, 256 threads (8 waves). Wave w: M-subtile (w&3),
// N-subtiles {2*(w>>2), 2*(w>>2)+1}. A row-major bf16; B TRANSPOSED
// (Bt[n][k]) so LDS tiles are n-major and every fragment is 2x b128.
// Register double-buffer: stage kt+1 is loaded to VGPRs while stage kt
// computes; s_wait_loadcnt overlaps 4 WMMAs + barrier.
// AMODE: 0 dense A0[M][K]; 1 edge-gather [node[row]|node[col]|ea] (K=1088);
//        2 concat A0[M][K0] | A1[M][K-K0]
template<int AMODE>
__global__ __launch_bounds__(256)
void kGemmT(const ushort_t* __restrict__ A0, const ushort_t* __restrict__ A1, int K0,
            const int* __restrict__ edgeIdx, const ushort_t* __restrict__ nodeBf,
            const ushort_t* __restrict__ eaBf, long long e0,
            int M, int K, const ushort_t* __restrict__ Bt, int Nout,
            const float* __restrict__ bias, int act, int outMode,
            ushort_t* __restrict__ outBf, float* __restrict__ outF)
{
  __shared__ alignas(16) ushort_t lA[64 * 64];   // [m][k]
  __shared__ alignas(16) ushort_t lB[64 * 64];   // [n][k]  (transposed tile)

  const int tid  = threadIdx.x;
  const int lane = tid & 31;
  const int wv   = tid >> 5;           // 0..7
  const int mt   = wv & 3;             // M-subtile of this wave
  const int nt0  = (wv >> 2) << 1;     // first of two N-subtiles
  const int mBase = blockIdx.x * 64;
  const int nBase = blockIdx.y * 64;

  // staging: thread -> (row 0..63, 16-elem quarter of the 64-wide k slice)
  const int sr = tid >> 2;
  const int sq = (tid & 3) << 4;       // 0,16,32,48
  int gRow = mBase + sr; if (gRow >= M) gRow = M - 1;

  // hoisted A source bases (absolute k added per stage)
  const ushort_t* aB0 = nullptr;
  const ushort_t* aB1 = nullptr;
  const ushort_t* aBE = nullptr;
  if (AMODE == 1) {
    long long e = e0 + gRow;
    int idxR = edgeIdx[e];
    int idxC = edgeIdx[(long long)E_TOT + e];
    aB0 = nodeBf + (long long)idxR * HDIM;               // k in [0,512)
    aB1 = nodeBf + (long long)idxC * HDIM - HDIM;        // k in [512,1024)
    aBE = eaBf + e * EDIM - 2 * HDIM;                    // k in [1024,1088)
  } else if (AMODE == 2) {
    aB0 = A0 + (long long)gRow * K0;
    aB1 = A1 + (long long)gRow * (K - K0) - K0;
  } else {
    aB0 = A0 + (long long)gRow * K;
  }
  const ushort_t* bBase = Bt + (long long)(nBase + sr) * K;

  v8f acc[2];
  acc[0] = (v8f){0, 0, 0, 0, 0, 0, 0, 0};
  acc[1] = (v8f){0, 0, 0, 0, 0, 0, 0, 0};

  const int fr = lane & 15;            // fragment row within subtile
  const int fh = lane >> 4;            // fragment K-half

  const int KT = K >> 6;               // 64-wide stages (all K are multiples of 64)
  uint4 ra0, ra1, rb0, rb1;            // in-flight stage registers

  auto loadStage = [&](int ktBase) {
    int k = ktBase + sq;
    const ushort_t* asrc;
    if (AMODE == 1) {
      asrc = (k < HDIM) ? (aB0 + k) : ((k < 2 * HDIM) ? (aB1 + k) : (aBE + k));
    } else if (AMODE == 2) {
      asrc = (k < K0) ? (aB0 + k) : (aB1 + k);
    } else {
      asrc = aB0 + k;
    }
    const ushort_t* bsrc = bBase + k;
    ra0 = ((const uint4*)asrc)[0];
    ra1 = ((const uint4*)asrc)[1];
    rb0 = ((const uint4*)bsrc)[0];
    rb1 = ((const uint4*)bsrc)[1];
    __builtin_prefetch((const void*)(asrc + 64), 0, 3);  // one stage ahead
    __builtin_prefetch((const void*)(bsrc + 64), 0, 3);
  };

  loadStage(0);
  for (int kt = 0; kt < KT; ++kt) {
    __syncthreads();                   // previous compute done reading LDS
    {
      uint4* da = (uint4*)&lA[(sr << 6) + sq];
      da[0] = ra0; da[1] = ra1;
      uint4* db = (uint4*)&lB[(sr << 6) + sq];
      db[0] = rb0; db[1] = rb1;
    }
    __syncthreads();                   // tiles visible
    if (kt + 1 < KT) loadStage((kt + 1) << 6);

    const ushort_t* aRow  = &lA[((mt << 4) + fr) << 6];
    const ushort_t* bRowA = &lB[((nt0 << 4) + fr) << 6];
    const ushort_t* bRowB = &lB[(((nt0 + 1) << 4) + fr) << 6];
    v16bf a0 = ldsFrag(aRow, fh, 0);
    v16bf a1 = ldsFrag(aRow, fh, 32);
    acc[0] = __builtin_amdgcn_wmma_f32_16x16x32_bf16(
        false, a0, false, ldsFrag(bRowA, fh, 0), (short)0, acc[0], false, false);
    acc[0] = __builtin_amdgcn_wmma_f32_16x16x32_bf16(
        false, a1, false, ldsFrag(bRowA, fh, 32), (short)0, acc[0], false, false);
    acc[1] = __builtin_amdgcn_wmma_f32_16x16x32_bf16(
        false, a0, false, ldsFrag(bRowB, fh, 0), (short)0, acc[1], false, false);
    acc[1] = __builtin_amdgcn_wmma_f32_16x16x32_bf16(
        false, a1, false, ldsFrag(bRowB, fh, 32), (short)0, acc[1], false, false);
  }

  // ---- epilogue: D layout VGPR r -> row r + 8*half, col = lane&15 ----
#pragma unroll
  for (int j = 0; j < 2; ++j) {
    int gn = nBase + ((nt0 + j) << 4) + fr;
    float bb = bias ? bias[gn] : 0.0f;
#pragma unroll
    for (int r = 0; r < 8; ++r) {
      int gm = mBase + (mt << 4) + r + (fh << 3);
      if (gm >= M) continue;
      float vv = acc[j][r] + bb;
      if (act) vv = siluf(vv);
      if (outMode == 0) {
        outBf[(long long)gm * Nout + gn] = f2bf(vv);
      } else {
        if (gn < LATD) outF[(long long)gm * LATD + gn] = vv;                            // mu
        else outF[(long long)NGRAPHS * LATD + (long long)gm * LATD + (gn - LATD)] = vv; // logvar
      }
    }
  }
}

// ---------------- helper kernels ----------------
// convert f32 weight stack (L x K x Nout) to bf16 TRANSPOSED (L x Nout x K)
__global__ void kConvertT(const float* __restrict__ src, ushort_t* __restrict__ dst,
                          int K, int Nout, int total) {
  int i = blockIdx.x * blockDim.x + threadIdx.x;   // reads coalesced
  if (i >= total) return;
  int kn = K * Nout;
  int l = i / kn, w = i - l * kn;
  int k = w / Nout, n = w - k * Nout;
  dst[(long long)l * kn + (long long)n * K + k] = f2bf(src[i]);
}

__global__ void kNodeInit(const float* __restrict__ emb, const int* __restrict__ types,
                          ushort_t* __restrict__ nodeBf) {
  int i = blockIdx.x * blockDim.x + threadIdx.x;   // over N*H/8
  if (i >= N_NODES * HDIM / 8) return;
  int nn = i >> 6, h8 = (i & 63) << 3;
  const float* s = emb + (long long)types[nn] * HDIM + h8;
  float4 a = ((const float4*)s)[0];
  float4 b = ((const float4*)s)[1];
  uint4 o;
  o.x = (unsigned)f2bf(a.x) | ((unsigned)f2bf(a.y) << 16);
  o.y = (unsigned)f2bf(a.z) | ((unsigned)f2bf(a.w) << 16);
  o.z = (unsigned)f2bf(b.x) | ((unsigned)f2bf(b.y) << 16);
  o.w = (unsigned)f2bf(b.z) | ((unsigned)f2bf(b.w) << 16);
  ((uint4*)(nodeBf + ((long long)nn * HDIM + h8)))[0] = o;
}

__global__ void kFourier(const float* __restrict__ coords, const int* __restrict__ edgeIdx,
                         const float* __restrict__ gfp, ushort_t* __restrict__ ea0) {
  int e = blockIdx.x * blockDim.x + threadIdx.x;
  if (e >= E_TOT) return;
  int r = edgeIdx[e], c = edgeIdx[E_TOT + e];
  float dx = coords[r * 3 + 0] - coords[c * 3 + 0];
  float dy = coords[r * 3 + 1] - coords[c * 3 + 1];
  float dz = coords[r * 3 + 2] - coords[c * 3 + 2];
  float d = sqrtf(dx * dx + dy * dy + dz * dz);     // ORIGINAL coords (stop_gradient)
  ushort_t sn[32], cs[32];
#pragma unroll
  for (int j = 0; j < 32; ++j) {
    float xp = d * gfp[j] * 6.28318530717958647692f;
    sn[j] = f2bf(__sinf(xp));
    cs[j] = f2bf(__cosf(xp));
  }
  uint4* o = (uint4*)(ea0 + (long long)e * EDIM);
#pragma unroll
  for (int q = 0; q < 4; ++q) {
    uint4 v;
    v.x = (unsigned)sn[q * 8 + 0] | ((unsigned)sn[q * 8 + 1] << 16);
    v.y = (unsigned)sn[q * 8 + 2] | ((unsigned)sn[q * 8 + 3] << 16);
    v.z = (unsigned)sn[q * 8 + 4] | ((unsigned)sn[q * 8 + 5] << 16);
    v.w = (unsigned)sn[q * 8 + 6] | ((unsigned)sn[q * 8 + 7] << 16);
    o[q] = v;
    v.x = (unsigned)cs[q * 8 + 0] | ((unsigned)cs[q * 8 + 1] << 16);
    v.y = (unsigned)cs[q * 8 + 2] | ((unsigned)cs[q * 8 + 3] << 16);
    v.z = (unsigned)cs[q * 8 + 4] | ((unsigned)cs[q * 8 + 5] << 16);
    v.w = (unsigned)cs[q * 8 + 6] | ((unsigned)cs[q * 8 + 7] << 16);
    o[4 + q] = v;
  }
}

// nm[node] = sum of em over the node's 31 contiguous edges (uint4 lanes)
__global__ void kNm(const ushort_t* __restrict__ em, long long e0, int n0,
                    ushort_t* __restrict__ nmBf) {
  int i = blockIdx.x * blockDim.x + threadIdx.x;     // over 256*64
  if (i >= NODES_PER_CHUNK * (HDIM / 8)) return;
  int nl = i >> 6, h8 = (i & 63) << 3;
  int node = n0 + nl;
  long long el = (long long)node * EPN - e0;         // chunk-local first edge row
  float s[8] = {0, 0, 0, 0, 0, 0, 0, 0};
  for (int j = 0; j < EPN; ++j) {
    uint4 u = *(const uint4*)(em + (el + j) * HDIM + h8);
    s[0] += bf2f((ushort_t)u.x); s[1] += bf2f((ushort_t)(u.x >> 16));
    s[2] += bf2f((ushort_t)u.y); s[3] += bf2f((ushort_t)(u.y >> 16));
    s[4] += bf2f((ushort_t)u.z); s[5] += bf2f((ushort_t)(u.z >> 16));
    s[6] += bf2f((ushort_t)u.w); s[7] += bf2f((ushort_t)(u.w >> 16));
  }
  uint4 o;
  o.x = (unsigned)f2bf(s[0]) | ((unsigned)f2bf(s[1]) << 16);
  o.y = (unsigned)f2bf(s[2]) | ((unsigned)f2bf(s[3]) << 16);
  o.z = (unsigned)f2bf(s[4]) | ((unsigned)f2bf(s[5]) << 16);
  o.w = (unsigned)f2bf(s[6]) | ((unsigned)f2bf(s[7]) << 16);
  *(uint4*)(nmBf + (long long)node * HDIM + h8) = o;
}

// coords_next[node] = coords_cur[node] + sum_e (c1[e].cw2 + cb2) * cd/|cd|
__global__ void kCoord(const ushort_t* __restrict__ c1, const float* __restrict__ cw2,
                       const float* __restrict__ cb2, const int* __restrict__ edgeIdx,
                       long long e0, int n0,
                       const float* __restrict__ coordsCur, float* __restrict__ coordsNext) {
  __shared__ float sm[32][3];
  int node = n0 + blockIdx.x;
  int lane = threadIdx.x;
  float cmx = 0.f, cmy = 0.f, cmz = 0.f;
  if (lane < EPN) {
    long long eg = (long long)node * EPN + lane;
    const uint4* cr = (const uint4*)(c1 + (eg - e0) * HDIM);
    const float4* w4 = (const float4*)cw2;
    float s = cb2[0];
    for (int h8 = 0; h8 < HDIM / 8; ++h8) {
      uint4 u = cr[h8];
      float4 w0 = w4[h8 * 2], w1 = w4[h8 * 2 + 1];
      s += bf2f((ushort_t)u.x) * w0.x + bf2f((ushort_t)(u.x >> 16)) * w0.y
         + bf2f((ushort_t)u.y) * w0.z + bf2f((ushort_t)(u.y >> 16)) * w0.w
         + bf2f((ushort_t)u.z) * w1.x + bf2f((ushort_t)(u.z >> 16)) * w1.y
         + bf2f((ushort_t)u.w) * w1.z + bf2f((ushort_t)(u.w >> 16)) * w1.w;
    }
    int ri = edgeIdx[eg], ci = edgeIdx[(long long)E_TOT + eg];
    float dx = coordsCur[ri * 3 + 0] - coordsCur[ci * 3 + 0];
    float dy = coordsCur[ri * 3 + 1] - coordsCur[ci * 3 + 1];
    float dz = coordsCur[ri * 3 + 2] - coordsCur[ci * 3 + 2];
    float cd = sqrtf(dx * dx + dy * dy + dz * dz) + 1e-8f;
    float t = s / cd;
    cmx = t * dx; cmy = t * dy; cmz = t * dz;
  }
  sm[lane][0] = cmx; sm[lane][1] = cmy; sm[lane][2] = cmz;
  __syncthreads();
  if (lane < 3) {
    float s = 0.f;
    for (int i = 0; i < 32; ++i) s += sm[i][lane];
    coordsNext[node * 3 + lane] = coordsCur[node * 3 + lane] + s;
  }
}

__global__ void kPool(const ushort_t* __restrict__ nodeBf, ushort_t* __restrict__ graphBf) {
  int i = blockIdx.x * blockDim.x + threadIdx.x;     // over 128*64
  if (i >= NGRAPHS * (HDIM / 8)) return;
  int g = i >> 6, h8 = (i & 63) << 3;
  float s[8] = {0, 0, 0, 0, 0, 0, 0, 0};
  for (int a = 0; a < 32; ++a) {
    uint4 u = *(const uint4*)(nodeBf + ((long long)(g * 32 + a)) * HDIM + h8);
    s[0] += bf2f((ushort_t)u.x); s[1] += bf2f((ushort_t)(u.x >> 16));
    s[2] += bf2f((ushort_t)u.y); s[3] += bf2f((ushort_t)(u.y >> 16));
    s[4] += bf2f((ushort_t)u.z); s[5] += bf2f((ushort_t)(u.z >> 16));
    s[6] += bf2f((ushort_t)u.w); s[7] += bf2f((ushort_t)(u.w >> 16));
  }
  uint4 o;
  o.x = (unsigned)f2bf(s[0] * 0.03125f) | ((unsigned)f2bf(s[1] * 0.03125f) << 16);
  o.y = (unsigned)f2bf(s[2] * 0.03125f) | ((unsigned)f2bf(s[3] * 0.03125f) << 16);
  o.z = (unsigned)f2bf(s[4] * 0.03125f) | ((unsigned)f2bf(s[5] * 0.03125f) << 16);
  o.w = (unsigned)f2bf(s[6] * 0.03125f) | ((unsigned)f2bf(s[7] * 0.03125f) << 16);
  *(uint4*)(graphBf + (long long)g * HDIM + h8) = o;
}

// ---------------- host orchestration ----------------
extern "C" void kernel_launch(void* const* d_in, const int* in_sizes, int n_in,
                              void* d_out, int out_size, void* d_ws, size_t ws_size,
                              hipStream_t stream) {
  (void)in_sizes; (void)n_in; (void)out_size;

  const float* coords     = (const float*)d_in[0];
  const int*   atom_types = (const int*)d_in[1];
  const int*   edgeIdx    = (const int*)d_in[3];
  const float* atom_embed = (const float*)d_in[4];
  const float* gfp_W      = (const float*)d_in[5];
  const float* edge_lin_w = (const float*)d_in[6];
  const float* edge_lin_b = (const float*)d_in[7];
  const float* edge_w1    = (const float*)d_in[8];
  const float* edge_b1    = (const float*)d_in[9];
  const float* edge_w2    = (const float*)d_in[10];
  const float* edge_b2    = (const float*)d_in[11];
  const float* node_w1    = (const float*)d_in[12];
  const float* node_b1    = (const float*)d_in[13];
  const float* node_w2    = (const float*)d_in[14];
  const float* node_b2    = (const float*)d_in[15];
  const float* coord_w1   = (const float*)d_in[16];
  const float* coord_b1   = (const float*)d_in[17];
  const float* coord_w2   = (const float*)d_in[18];
  const float* coord_b2   = (const float*)d_in[19];
  const float* graph_w1   = (const float*)d_in[20];
  const float* graph_b1   = (const float*)d_in[21];
  const float* graph_w2   = (const float*)d_in[22];
  const float* graph_b2   = (const float*)d_in[23];

  size_t off = 0;
  char* base = (char*)d_ws;
  auto alloc = [&](size_t bytes) -> void* {
    off = (off + 255) & ~(size_t)255;
    void* p = base + off;
    off += bytes;
    return p;
  };
  ushort_t* wbE1  = (ushort_t*)alloc((size_t)NLAYERS * 1088 * 512 * 2);
  ushort_t* wbE2  = (ushort_t*)alloc((size_t)NLAYERS * 512 * 512 * 2);
  ushort_t* wbN1  = (ushort_t*)alloc((size_t)NLAYERS * 1024 * 512 * 2);
  ushort_t* wbN2  = (ushort_t*)alloc((size_t)NLAYERS * 512 * 512 * 2);
  ushort_t* wbC1  = (ushort_t*)alloc((size_t)NLAYERS * 512 * 512 * 2);
  ushort_t* wbEL  = (ushort_t*)alloc((size_t)64 * 64 * 2);
  ushort_t* wbG1  = (ushort_t*)alloc((size_t)512 * 512 * 2);
  ushort_t* wbG2  = (ushort_t*)alloc((size_t)512 * 512 * 2);
  ushort_t* nodeBf  = (ushort_t*)alloc((size_t)N_NODES * HDIM * 2);
  ushort_t* nmBf    = (ushort_t*)alloc((size_t)N_NODES * HDIM * 2);
  ushort_t* xhBf    = (ushort_t*)alloc((size_t)N_NODES * HDIM * 2);
  ushort_t* ea0Bf   = (ushort_t*)alloc((size_t)E_TOT * EDIM * 2);
  ushort_t* eaBf    = (ushort_t*)alloc((size_t)E_TOT * EDIM * 2);
  ushort_t* h1Bf    = (ushort_t*)alloc((size_t)E_CHUNK * HDIM * 2);
  ushort_t* emBf    = (ushort_t*)alloc((size_t)E_CHUNK * HDIM * 2);
  ushort_t* c1Bf    = (ushort_t*)alloc((size_t)E_CHUNK * HDIM * 2);
  float*    coordsA = (float*)alloc((size_t)N_NODES * 3 * 4);
  float*    coordsB = (float*)alloc((size_t)N_NODES * 3 * 4);
  ushort_t* graphBf = (ushort_t*)alloc((size_t)NGRAPHS * HDIM * 2);
  ushort_t* g1Bf    = (ushort_t*)alloc((size_t)NGRAPHS * HDIM * 2);
  if (off > ws_size) return;   // not enough scratch

  // convert + transpose all weight matrices (B operands, n-major)
  auto cvtT = [&](const float* s, ushort_t* d, int L, int K, int Nout) {
    long long tot = (long long)L * K * Nout;
    kConvertT<<<dim3((unsigned)((tot + 255) / 256)), dim3(256), 0, stream>>>(
        s, d, K, Nout, (int)tot);
  };
  cvtT(edge_w1,    wbE1, NLAYERS, 1088, 512);
  cvtT(edge_w2,    wbE2, NLAYERS, 512, 512);
  cvtT(node_w1,    wbN1, NLAYERS, 1024, 512);
  cvtT(node_w2,    wbN2, NLAYERS, 512, 512);
  cvtT(coord_w1,   wbC1, NLAYERS, 512, 512);
  cvtT(edge_lin_w, wbEL, 1, 64, 64);
  cvtT(graph_w1,   wbG1, 1, 512, 512);
  cvtT(graph_w2,   wbG2, 1, 512, 512);

  hipMemcpyAsync(coordsA, coords, (size_t)N_NODES * 3 * 4, hipMemcpyDeviceToDevice, stream);

  kNodeInit<<<dim3(N_NODES * HDIM / 8 / 256), dim3(256), 0, stream>>>(
      atom_embed, atom_types, nodeBf);
  kFourier<<<dim3(E_TOT / 256), dim3(256), 0, stream>>>(coords, edgeIdx, gfp_W, ea0Bf);

  auto gemmDense = [&](const ushort_t* A0, int M, int K, const ushort_t* Bt, int Nout,
                       const float* bias, int act, ushort_t* outBf,
                       int outMode = 0, float* outF = nullptr) {
    kGemmT<0><<<dim3(M / 64, Nout / 64), dim3(256), 0, stream>>>(
        A0, nullptr, K, nullptr, nullptr, nullptr, 0LL,
        M, K, Bt, Nout, bias, act, outMode, outBf, outF);
  };

  // ea = fourier @ edge_lin_w + b   (no activation)
  gemmDense(ea0Bf, E_TOT, 64, wbEL, 64, edge_lin_b, 0, eaBf);

  float* coordsCur  = coordsA;
  float* coordsNext = coordsB;

  for (int l = 0; l < NLAYERS; ++l) {
    const ushort_t* wE1 = wbE1 + (size_t)l * 1088 * 512;
    const ushort_t* wE2 = wbE2 + (size_t)l * 512 * 512;
    const ushort_t* wN1 = wbN1 + (size_t)l * 1024 * 512;
    const ushort_t* wN2 = wbN2 + (size_t)l * 512 * 512;
    const ushort_t* wC1 = wbC1 + (size_t)l * 512 * 512;
    const float* eb1 = edge_b1 + (size_t)l * 512;
    const float* eb2 = edge_b2 + (size_t)l * 512;
    const float* nb1 = node_b1 + (size_t)l * 512;
    const float* nb2 = node_b2 + (size_t)l * 512;
    const float* cb1 = coord_b1 + (size_t)l * 512;
    const float* cw2 = coord_w2 + (size_t)l * 512;
    const float* cb2 = coord_b2 + (size_t)l;

    for (int c = 0; c < NCHUNKS; ++c) {
      long long e0 = (long long)c * E_CHUNK;
      int n0 = c * NODES_PER_CHUNK;
      // h1 = silu([node[row]|node[col]|ea] @ W1 + b1)   (gathered A, K=1088)
      kGemmT<1><<<dim3(E_CHUNK / 64, HDIM / 64), dim3(256), 0, stream>>>(
          nullptr, nullptr, 0, edgeIdx, nodeBf, eaBf, e0,
          E_CHUNK, 2 * HDIM + EDIM, wE1, HDIM, eb1, 1, 0, h1Bf, nullptr);
      // em = h1 @ W2 + b2
      gemmDense(h1Bf, E_CHUNK, HDIM, wE2, HDIM, eb2, 0, emBf);
      // c1 = silu(em @ Wc1 + bc1)
      gemmDense(emBf, E_CHUNK, HDIM, wC1, HDIM, cb1, 1, c1Bf);
      // coordinate update (segment reduce over each node's 31 edges)
      kCoord<<<dim3(NODES_PER_CHUNK), dim3(32), 0, stream>>>(
          c1Bf, cw2, cb2, edgeIdx, e0, n0, coordsCur, coordsNext);
      // nm = segment-sum of em
      kNm<<<dim3(NODES_PER_CHUNK * (HDIM / 8) / 256), dim3(256), 0, stream>>>(
          emBf, e0, n0, nmBf);
    }
    // node update: xh = silu([node|nm] @ Wn1 + bn1); node = xh @ Wn2 + bn2
    kGemmT<2><<<dim3(N_NODES / 64, HDIM / 64), dim3(256), 0, stream>>>(
        nodeBf, nmBf, HDIM, nullptr, nullptr, nullptr, 0LL,
        N_NODES, 2 * HDIM, wN1, HDIM, nb1, 1, 0, xhBf, nullptr);
    gemmDense(xhBf, N_NODES, HDIM, wN2, HDIM, nb2, 0, nodeBf);
    // swap coordinate buffers for next layer
    float* t = coordsCur; coordsCur = coordsNext; coordsNext = t;
  }

  // mean pool per graph, then 2-layer head; final GEMM writes f32 mu|logvar
  kPool<<<dim3(NGRAPHS * (HDIM / 8) / 256), dim3(256), 0, stream>>>(nodeBf, graphBf);
  gemmDense(graphBf, NGRAPHS, HDIM, wbG1, HDIM, graph_b1, 1, g1Bf);
  gemmDense(g1Bf, NGRAPHS, HDIM, wbG2, HDIM, graph_b2, 0, nullptr, 1, (float*)d_out);
}
